// STU_7928509629254
// MI455X (gfx1250) — matrix operand order repaired and checked by reading, not compiled, not run on hardware
//
#include <hip/hip_runtime.h>
#include <hip/hip_bf16.h>

// ---------------------------------------------------------------------------
// STU forward for MI455X (gfx1250).
// Math reduction: U_plus + U_minus == causal depthwise conv with only EVEN
// lags, taps = 2*phi_proj[2j].  Pipeline: bf16 convert (+one-time transpose
// of GEMM B operands) -> WMMA GEMM (double-buffered LDS, async-to-LDS tile
// staging) -> even-lag depthwise conv with packed f32 FMAs.  No FFT.
// ---------------------------------------------------------------------------

typedef __bf16 bf16_t;
typedef __attribute__((ext_vector_type(16))) __bf16 v16bf;
typedef __attribute__((ext_vector_type(8)))  __bf16 v8bf;
typedef __attribute__((ext_vector_type(8)))  float  v8f;
typedef __attribute__((ext_vector_type(2)))  float  f32x2;

#if defined(__HIP_DEVICE_COMPILE__) && \
    __has_builtin(__builtin_amdgcn_global_load_async_to_lds_b128)
#define STU_ASYNC_LDS 1
#else
#define STU_ASYNC_LDS 0
#endif

#if STU_ASYNC_LDS
// Builtin prototype (from compiler diagnostic): 128-bit int-vector pointers,
// global (AS1) source, LDS (AS3) destination, then imm offset + cpol.
typedef int stu_v4i __attribute__((vector_size(4 * sizeof(int))));
typedef __attribute__((address_space(1))) stu_v4i* stu_gp128;
typedef __attribute__((address_space(3))) stu_v4i* stu_lp128;

__device__ __forceinline__ void stu_async_cp16(const bf16_t* g, bf16_t* l) {
    __builtin_amdgcn_global_load_async_to_lds_b128(
        (stu_gp128)(g), (stu_lp128)(l), 0, 0);
}
__device__ __forceinline__ void stu_wait_async0() {
#if __has_builtin(__builtin_amdgcn_s_wait_asynccnt)
    __builtin_amdgcn_s_wait_asynccnt(0);
#else
    asm volatile("s_wait_asynccnt 0x0" ::: "memory");
#endif
}
#endif

__device__ __forceinline__ bf16_t f2bf(float f) {
    unsigned u = __float_as_uint(f);
    unsigned r = (u + 0x7FFFu + ((u >> 16) & 1u)) >> 16;   // round-to-nearest-even
    unsigned short s = (unsigned short)r;
    bf16_t b;
    __builtin_memcpy(&b, &s, 2);
    return b;
}

// ------------------------------ conversions --------------------------------

__global__ __launch_bounds__(256)
void stu_cvt_f32_bf16(const float* __restrict__ in, bf16_t* __restrict__ out, int n) {
    int i = blockIdx.x * blockDim.x + threadIdx.x;
    int stride = gridDim.x * blockDim.x;
    for (; i < n; i += stride) out[i] = f2bf(in[i]);
}

// Transposing convert: in[R][C] f32 (row-major) -> out[C][R] bf16.
__global__ __launch_bounds__(256)
void stu_cvt_f32_bf16_T(const float* __restrict__ in, bf16_t* __restrict__ out,
                        int R, int C) {
    __shared__ bf16_t tile[32][33];
    const int bx = blockIdx.x * 32;   // input col base
    const int by = blockIdx.y * 32;   // input row base
    const int tx = threadIdx.x & 31;
    const int ty = threadIdx.x >> 5;  // 0..7
#pragma unroll
    for (int r = ty; r < 32; r += 8)
        tile[r][tx] = f2bf(in[(size_t)(by + r) * C + bx + tx]);
    __syncthreads();
#pragma unroll
    for (int r = ty; r < 32; r += 8)
        out[(size_t)(bx + r) * R + by + tx] = tile[tx][r];
}

// out[j*Kp+k] = bf16( 2 * phi[(2j)*Kp + k] ), j in [0, Jh)
__global__ __launch_bounds__(256)
void stu_cvt_phi_even_x2(const float* __restrict__ phi, bf16_t* __restrict__ out,
                         int Jh, int Kp) {
    int i = blockIdx.x * blockDim.x + threadIdx.x;
    int stride = gridDim.x * blockDim.x;
    int n = Jh * Kp;
    for (; i < n; i += stride) {
        int j = i / Kp, k = i - j * Kp;
        out[i] = f2bf(2.0f * phi[(size_t)(2 * j) * Kp + k]);
    }
}

// ------------------------------ WMMA GEMM ----------------------------------
// C[M,N] = A[M,K] * Bt[N,K]^T, bf16 in, f32 out.  B PRE-TRANSPOSED.  Strides
// are template constants so row offsets fold into instruction offsets.
// Tile 128x128x32, 256 thr, 8 waves, each wave 32x64 (2x4 WMMA accums).
// Double-buffered LDS; next K-tile staged with GLOBAL_LOAD_ASYNC_TO_LDS_B128
// (ASYNCcnt) when available, else via registers.

#define GBM 128
#define GBN 128
#define GBK 32
#define GLS 48   // LDS row stride in bf16 (32 data + 16 pad -> 96B; 32B aligned)

template <int LDA, int LDBT, int LDC>
__global__ __launch_bounds__(256)
void stu_gemm_bf16_wmma(const bf16_t* __restrict__ A,
                        const bf16_t* __restrict__ Bt,   // [N][K]
                        float*        __restrict__ C,
                        int K) {
    __shared__ bf16_t sA[2][GBM * GLS];
    __shared__ bf16_t sB[2][GBN * GLS];

    const int tid  = threadIdx.x;
    const int lane = tid & 31;
    const int wave = tid >> 5;
    const int wm   = (wave & 3) * 32;   // wave M offset in tile
    const int wn   = (wave >> 2) * 64;  // wave N offset in tile

    const int bm = blockIdx.y * GBM;
    const int bn = blockIdx.x * GBN;

    v8f zero = {};
    v8f acc[2][4];
#pragma unroll
    for (int i = 0; i < 2; i++)
#pragma unroll
        for (int j = 0; j < 4; j++) acc[i][j] = zero;

    // Both tiles are 128 rows x 32 bf16: thread -> row tr (0..127), col (tid&1)*16
    const int tr = tid >> 1, tc = (tid & 1) * 16;
    const bf16_t* gA = A  + (size_t)(bm + tr) * LDA  + tc;
    const bf16_t* gB = Bt + (size_t)(bn + tr) * LDBT + tc;
    const int lofs = tr * GLS + tc;

    // prologue: stage K-tile 0 into buffer 0
#if STU_ASYNC_LDS
    {
        stu_async_cp16(gA,     sA[0] + lofs);
        stu_async_cp16(gA + 8, sA[0] + lofs + 8);
        stu_async_cp16(gB,     sB[0] + lofs);
        stu_async_cp16(gB + 8, sB[0] + lofs + 8);
        stu_wait_async0();
    }
#else
    {
        uint4 a0 = ((const uint4*)gA)[0];
        uint4 a1 = ((const uint4*)gA)[1];
        uint4 b0 = ((const uint4*)gB)[0];
        uint4 b1 = ((const uint4*)gB)[1];
        *(uint4*)(sA[0] + lofs)     = a0;
        *(uint4*)(sA[0] + lofs + 8) = a1;
        *(uint4*)(sB[0] + lofs)     = b0;
        *(uint4*)(sB[0] + lofs + 8) = b1;
    }
#endif

    // fragment lane mapping (ISA 7.12.2)
    const int fr    = lane & 15;
    const int akoff = (lane < 16) ? 0 : 16;   // A: K 0..15 vs 16..31
    const int bkoff = (lane < 16) ? 0 : 16;   // B: K 0..15 vs 16..31

    int p = 0;
    for (int k0 = 0; k0 < K; k0 += GBK) {
        const bool has_next = (k0 + GBK < K);

#if STU_ASYNC_LDS
        if (has_next) {   // stage next tile straight into LDS buffer p^1
            const bf16_t* nA = gA + k0 + GBK;
            const bf16_t* nB = gB + k0 + GBK;
            bf16_t* dA = sA[p ^ 1] + lofs;
            bf16_t* dB = sB[p ^ 1] + lofs;
            stu_async_cp16(nA,     dA);
            stu_async_cp16(nA + 8, dA + 8);
            stu_async_cp16(nB,     dB);
            stu_async_cp16(nB + 8, dB + 8);
            if (k0 + 2 * GBK < K) {
                __builtin_prefetch(gA + k0 + 2 * GBK, 0, 0);
                __builtin_prefetch(gB + k0 + 2 * GBK, 0, 0);
            }
        }
#else
        uint4 a0, a1, b0, b1;
        if (has_next) {
            const bf16_t* nA = gA + k0 + GBK;
            const bf16_t* nB = gB + k0 + GBK;
            a0 = ((const uint4*)nA)[0];
            a1 = ((const uint4*)nA)[1];
            b0 = ((const uint4*)nB)[0];
            b1 = ((const uint4*)nB)[1];
            if (k0 + 2 * GBK < K) {
                __builtin_prefetch(gA + k0 + 2 * GBK, 0, 0);
                __builtin_prefetch(gB + k0 + 2 * GBK, 0, 0);
            }
        }
#endif
        __syncthreads();   // current buffer p fully staged & published

        v16bf av[2], bv[4];
#pragma unroll
        for (int mi = 0; mi < 2; mi++) {
            const bf16_t* q = sA[p] + (wm + mi * 16 + fr) * GLS + akoff;
            union { v16bf v; v8bf h[2]; } u;
            u.h[0] = *(const v8bf*)(q);
            u.h[1] = *(const v8bf*)(q + 8);
            av[mi] = u.v;
        }
#pragma unroll
        for (int ni = 0; ni < 4; ni++)   // 32B aligned contiguous read
            bv[ni] = *(const v16bf*)(sB[p] + (wn + ni * 16 + fr) * GLS + bkoff);

#pragma unroll
        for (int mi = 0; mi < 2; mi++)
#pragma unroll
            for (int ni = 0; ni < 4; ni++)
                acc[mi][ni] = __builtin_amdgcn_wmma_f32_16x16x32_bf16(
                    false, av[mi], false, bv[ni], (short)0, acc[mi][ni],
                    false, false);

#if STU_ASYNC_LDS
        if (has_next) stu_wait_async0();  // p^1 resident before next barrier
#else
        if (has_next) {
            bf16_t* dA = sA[p ^ 1] + lofs;
            bf16_t* dB = sB[p ^ 1] + lofs;
            *(uint4*)(dA)     = a0;
            *(uint4*)(dA + 8) = a1;
            *(uint4*)(dB)     = b0;
            *(uint4*)(dB + 8) = b1;
        }
#endif
        p ^= 1;
    }

    // C/D layout: VGPR r, lanes 0-15 -> (M = base+r, N = lane); lanes 16-31 -> +8
    const int colr = lane & 15;
    const int rhi  = (lane < 16) ? 0 : 8;
#pragma unroll
    for (int mi = 0; mi < 2; mi++)
#pragma unroll
        for (int ni = 0; ni < 4; ni++) {
            const int col  = bn + wn + ni * 16 + colr;
            const int row0 = bm + wm + mi * 16 + rhi;
            float* cp = C + (size_t)row0 * LDC + col;
#pragma unroll
            for (int r = 0; r < 8; r++)
                cp[(size_t)r * LDC] = acc[mi][ni][r];   // r*LDC folds to imm offset
        }
}

// --------------------------- even-lag depthwise conv -----------------------
// out[b,t,d] = sum_{j=0}^{t/2} w[j,d] * u[b, t-2j, d]
// Channel-pair f32x2 lanes -> packed FMAs (v_pk_fma_f32) fed by ds_load_b64.
// Block = (128 t) x (64 ch) x (1 b); lag chunks of 32; 192-row f32x2 window.

#define CVT 128   // t per block
#define CVP 32    // f32x2 channel-pairs per block (64 scalar channels)
#define CVJ 32    // lags per chunk
#define CVW 192   // LDS window rows (>= CVT + 2*(CVJ-1) = 190)

template <int L, int D, int J>
__global__ __launch_bounds__(256)
void stu_conv_even_lag(const float* __restrict__ u,   // [B][L][D]
                       const float* __restrict__ w,   // [J][D]
                       float* __restrict__ out) {     // [B][L][D]
    __shared__ f32x2 su[CVW * CVP];   // 48 KB
    __shared__ f32x2 sw[CVJ * CVP];   //  8 KB

    constexpr int Dh = D >> 1;

    const int tid = threadIdx.x;
    const int c2  = tid & (CVP - 1);   // channel pair (== lane)
    const int tg  = tid >> 5;          // wave id: 16 t values each
    const int t0  = blockIdx.x * CVT;
    const int d0  = blockIdx.y * (2 * CVP);
    const int b   = blockIdx.z;

    const f32x2* ub2 = (const f32x2*)(u + (size_t)b * L * D + d0);
    const f32x2* w2  = (const f32x2*)(w + d0);

    const f32x2 zv = {0.0f, 0.0f};
    f32x2 acc[16];
#pragma unroll
    for (int i = 0; i < 16; i++) acc[i] = zv;

    const int jmax = min(J - 1, (t0 + CVT - 1) >> 1);

    for (int j0 = 0; j0 <= jmax; j0 += CVJ) {
        const int tw0 = t0 - 2 * j0 - 2 * (CVJ - 1);  // window start (may be <0)

        // fill input window: 192 x 32 f32x2, coalesced, zero outside [0,L)
#pragma unroll
        for (int r = 0; r < (CVW * CVP) / 256; r++) {   // 24 iters
            int e  = r * 256 + tid;
            int tt = e >> 5;            // / CVP
            int cc = e & (CVP - 1);
            int t  = tw0 + tt;
            su[e] = (t >= 0 && t < L) ? ub2[(size_t)t * Dh + cc] : zv;
        }
        // fill filter chunk: 32 x 32 f32x2
#pragma unroll
        for (int r = 0; r < (CVJ * CVP) / 256; r++) {   // 4 iters
            int e  = r * 256 + tid;
            int jj = e >> 5;
            int cc = e & (CVP - 1);
            int j  = j0 + jj;
            sw[e] = (j < J) ? w2[(size_t)j * Dh + cc] : zv;
        }
        __syncthreads();

        const int tb = tg * 16;
#pragma unroll 4
        for (int jj = 0; jj < CVJ; jj++) {
            f32x2 wv = sw[jj * CVP + c2];
            int base = tb + 2 * (CVJ - 1) - 2 * jj;  // su row for i=0
#pragma unroll
            for (int i = 0; i < 16; i++)
                acc[i] += wv * su[(base + i) * CVP + c2];   // v_pk_fma_f32
        }
        __syncthreads();
    }

    f32x2* ob2 = (f32x2*)(out + (size_t)b * L * D + d0);
    const int tout = t0 + tg * 16;
#pragma unroll
    for (int i = 0; i < 16; i++)
        ob2[(size_t)(tout + i) * Dh + c2] = acc[i];
}

// ------------------------------- launcher ----------------------------------

extern "C" void kernel_launch(void* const* d_in, const int* in_sizes, int n_in,
                              void* d_out, int out_size, void* d_ws, size_t ws_size,
                              hipStream_t stream) {
    const float* x   = (const float*)d_in[0];   // [B,L,D]
    const float* phi = (const float*)d_in[1];   // [L,Kp]
    const float* Mi  = (const float*)d_in[2];   // [D,D]
    const float* Mf  = (const float*)d_in[3];   // [Kp,D]
    // d_in[4] = n (FFT length) -- unused: conv reduction makes it irrelevant.

    const int B = 4, L = 2048, D = 1024, Kp = 256;
    const int Jh = L / 2;        // 1024 even-lag taps
    const int BL = B * L;        // 8192 GEMM1 rows

    char* ws = (char*)d_ws;
    bf16_t* x_bf   = (bf16_t*)ws;  ws += (size_t)BL * D * sizeof(bf16_t);
    bf16_t* MiT    = (bf16_t*)ws;  ws += (size_t)D * D * sizeof(bf16_t);   // [N][K]
    bf16_t* phiE   = (bf16_t*)ws;  ws += (size_t)Jh * Kp * sizeof(bf16_t);
    bf16_t* MfT    = (bf16_t*)ws;  ws += (size_t)Kp * D * sizeof(bf16_t);  // [N][K]
    float*  x_proj = (float*)ws;   ws += (size_t)BL * D * sizeof(float);
    float*  w_half = (float*)ws;   ws += (size_t)Jh * D * sizeof(float);

    // 1) precision conversions; GEMM B operands transposed once here
    stu_cvt_f32_bf16<<<2048, 256, 0, stream>>>(x, x_bf, BL * D);
    stu_cvt_f32_bf16_T<<<dim3(D / 32, D / 32), 256, 0, stream>>>(Mi, MiT, D, D);
    stu_cvt_f32_bf16_T<<<dim3(D / 32, Kp / 32), 256, 0, stream>>>(Mf, MfT, Kp, D);
    stu_cvt_phi_even_x2<<<256, 256, 0, stream>>>(phi, phiE, Jh, Kp);

    // 2) x_proj = x @ M_inputs     [8192 x 1024] * [1024 x 1024]
    stu_gemm_bf16_wmma<1024, 1024, 1024><<<dim3(D / GBN, BL / GBM), 256, 0, stream>>>(
        x_bf, MiT, x_proj, D);

    // 3) w_half = 2 * phi[even] @ M_filters   [1024 x 256] * [256 x 1024]
    stu_gemm_bf16_wmma<256, 256, 1024><<<dim3(D / GBN, Jh / GBM), 256, 0, stream>>>(
        phiE, MfT, w_half, Kp);

    // 4) even-lag causal depthwise conv -> output
    stu_conv_even_lag<2048, 1024, 1024><<<dim3(L / CVT, D / (2 * CVP), B), 256, 0, stream>>>(
        x_proj, w_half, (float*)d_out);
}
